// DiscRNNGrammar_58987080843461
// MI455X (gfx1250) — compile-verified, matrix operand order checked by reading, not run on hardware
//
#include <hip/hip_runtime.h>
#include <hip/hip_bf16.h>
#include <math.h>

// ---------------- model constants (match reference) ----------------
#define BB    128
#define SS    16
#define KK    4
#define TWW   64          // S*K
#define TT    96          // S*(K+2)
#define DD    256
#define HH    256
#define LL    2
#define AAN   66          // NNT + 2
#define AAPAD 80          // AAN padded to multiple of 16
#define GG    1024        // 4*H gates
#define DMAXX 23          // S+K+3
#define WDD   128
#define PDD   32
#define NTDD  128
#define ADD_  64

typedef __attribute__((ext_vector_type(16))) _Float16 v16h;
typedef __attribute__((ext_vector_type(8)))  _Float16 v8h;
typedef __attribute__((ext_vector_type(8)))  float    v8f;

#define WMMA_F16(acc, a, b) \
  (acc) = __builtin_amdgcn_wmma_f32_16x16x32_f16(false, (a), false, (b), (short)0, (acc), false, false)

__device__ __forceinline__ float sigmf_(float x) { return 1.0f / (1.0f + __expf(-x)); }

// ---- A fragment (16x32 f16) from f32 rows: lane&15 = M; lane>>4 = K-half ---
__device__ __forceinline__ v16h frag_a_f32(const float* __restrict__ base, long rowStride,
                                           int m0, int k0, int lane) {
  int m = lane & 15;
  int sel = (lane >> 4) & 1;
  const float* p = base + (long)(m0 + m) * rowStride + k0 + sel * 8;
  v8f lo = *(const v8f*)(p);
  v8f hi = *(const v8f*)(p + 16);
  v16h a;
#pragma unroll
  for (int j = 0; j < 8; ++j) a[j] = (_Float16)lo[j];
#pragma unroll
  for (int j = 0; j < 8; ++j) a[8 + j] = (_Float16)hi[j];
  return a;
}

// ---- A fragment from LDS (already f16) ------------------------------------
__device__ __forceinline__ v16h frag_a_lds(const _Float16* __restrict__ sh, int k0, int lane) {
  int m = lane & 15;
  int sel = (lane >> 4) & 1;
  const _Float16* p = sh + m * DD + k0 + sel * 8;
  v8h lo = *(const v8h*)(p);
  v8h hi = *(const v8h*)(p + 16);
  return __builtin_shufflevector(lo, hi, 0, 1, 2, 3, 4, 5, 6, 7, 8, 9, 10, 11, 12, 13, 14, 15);
}

// ---- B fragment from pre-converted f16 weight (N x K row-major): W^T tile --
__device__ __forceinline__ v16h frag_b_f16(const _Float16* __restrict__ W, int ldw,
                                           int n0, int k0, int lane) {
  int n = n0 + (lane & 15);
  int sel = (lane >> 4) & 1;
  const _Float16* p = W + (long)n * ldw + k0 + sel * 8;
  v8h lo = *(const v8h*)(p);
  v8h hi = *(const v8h*)(p + 16);
  return __builtin_shufflevector(lo, hi, 0, 1, 2, 3, 4, 5, 6, 7, 8, 9, 10, 11, 12, 13, 14, 15);
}

// 16 gate tiles per wave: n0 = q*256 + wave*64 + s*16
__device__ __forceinline__ void mma_16tiles(v8f acc[4][4], v16h a,
                                            const _Float16* __restrict__ Wl,
                                            int ldw, int wave, int k0, int lane) {
#pragma unroll
  for (int q = 0; q < 4; ++q)
#pragma unroll
    for (int s = 0; s < 4; ++s) {
      int n0 = q * 256 + wave * 64 + s * 16;
      v16h b = frag_b_f16(Wl, ldw, n0, k0, lane);
      WMMA_F16(acc[q][s], a, b);
    }
}

// ===========================================================================
// Fused 2-layer LSTM step. One workgroup = 16 batch rows, 4 waves.
// Wave w owns hidden cols [w*64, w*64+64): i/f/g/o of a hidden unit live in
// the same lane across acc[q][.], so gating needs no cross-lane traffic.
// ===========================================================================
__global__ __launch_bounds__(128) void rnng_lstm_step(
    const float* __restrict__ x, long xStride,
    const _Float16* __restrict__ Wih,  // (L,1024,256) f16
    const _Float16* __restrict__ Whh,  // (L,1024,256) f16
    const float* __restrict__ bias,    // (L,1024)
    const float* __restrict__ hIn, long hInStride,
    const float* __restrict__ cIn, long cInStride,
    float* __restrict__ hOut, long hOutStride,
    float* __restrict__ cOut, long cOutStride,
    float* __restrict__ outLast,       // (M,H) last-layer h, or nullptr
    float* __restrict__ topOut,        // (M,H) stack-top capture, or nullptr
    const float* __restrict__ topSrc, long topStride) {
  __shared__ _Float16 sH[16 * DD];  // layer-0 output -> layer-1 input
  const int m0 = blockIdx.x * 16;
  const int tid = threadIdx.x;
  const int wave = tid >> 5;
  const int lane = tid & 31;

  if (topOut != nullptr) {  // read-before-write top capture (stack scan)
    for (int i = tid; i < 16 * HH; i += 128) {
      int r = i >> 8, h = i & 255;
      topOut[(long)(m0 + r) * HH + h] = topSrc[(long)(m0 + r) * topStride + h];
    }
  }

  for (int l = 0; l < LL; ++l) {
    v8f acc[4][4];
    v8f z = {0.f, 0.f, 0.f, 0.f, 0.f, 0.f, 0.f, 0.f};
#pragma unroll
    for (int q = 0; q < 4; ++q)
#pragma unroll
      for (int s = 0; s < 4; ++s) acc[q][s] = z;

    const _Float16* Wl_ih = Wih + (long)l * GG * DD;
    const _Float16* Wl_hh = Whh + (long)l * GG * HH;

    // input contribution (branch hoisted out of the k-loop)
    if (l == 0) {
      for (int k0 = 0; k0 < DD; k0 += 32) {
        v16h a = frag_a_f32(x, xStride, m0, k0, lane);
        mma_16tiles(acc, a, Wl_ih, DD, wave, k0, lane);
      }
    } else {
      for (int k0 = 0; k0 < DD; k0 += 32) {
        v16h a = frag_a_lds(sH, k0, lane);
        mma_16tiles(acc, a, Wl_ih, DD, wave, k0, lane);
      }
    }
    // recurrent contribution: h_prev @ Whh^T
    for (int k0 = 0; k0 < HH; k0 += 32) {
      v16h a = frag_a_f32(hIn + (long)l * HH, hInStride, m0, k0, lane);
      mma_16tiles(acc, a, Wl_hh, HH, wave, k0, lane);
    }

    __syncthreads();  // all hIn/sH reads done before in-place state writes

    // gates + state update; C/D layout: acc[r] -> M = r + (lane>=16 ? 8 : 0)
    const int rowAdd = (lane >= 16) ? 8 : 0;
    const int colL = lane & 15;
#pragma unroll
    for (int s = 0; s < 4; ++s) {
      int h = wave * 64 + s * 16 + colL;
      float bi = bias[l * GG + h];
      float bf = bias[l * GG + 256 + h];
      float bg = bias[l * GG + 512 + h];
      float bo = bias[l * GG + 768 + h];
#pragma unroll
      for (int r = 0; r < 8; ++r) {
        int row = m0 + r + rowAdd;
        float iv = acc[0][s][r] + bi;
        float fv = acc[1][s][r] + bf;
        float gv = acc[2][s][r] + bg;
        float ov = acc[3][s][r] + bo;
        float cp = cIn[(long)row * cInStride + l * HH + h];
        float cn = sigmf_(fv) * cp + sigmf_(iv) * tanhf(gv);
        float hn = sigmf_(ov) * tanhf(cn);
        cOut[(long)row * cOutStride + l * HH + h] = cn;
        hOut[(long)row * hOutStride + l * HH + h] = hn;
        if (l == 0)
          sH[(r + rowAdd) * DD + h] = (_Float16)hn;
        else if (outLast != nullptr)
          outLast[(long)row * HH + h] = hn;
      }
    }
    __syncthreads();  // sH ready for layer 1
  }
}

// ===========================================================================
// Generic Out = act(A @ W^T + bias).  A: (M,K) f32; W16: (Npad,K) f16, rows
// >= N zero-padded.  k-outer so one A fragment feeds all of a wave's tiles.
// ===========================================================================
__global__ __launch_bounds__(128) void rnng_gemm_bias_act(
    const float* __restrict__ A, long aStride, int Kdim,
    const _Float16* __restrict__ W16, int N, int Npad,
    const float* __restrict__ bias,
    float* __restrict__ Out, long oStride, int act) {
  const int m0 = blockIdx.x * 16;
  const int tid = threadIdx.x;
  const int wave = tid >> 5;
  const int lane = tid & 31;
  const int nTiles = Npad >> 4;  // <= 16 for all uses here

  v8f acc[4];
  v8f z = {0.f, 0.f, 0.f, 0.f, 0.f, 0.f, 0.f, 0.f};
#pragma unroll
  for (int i = 0; i < 4; ++i) acc[i] = z;

  for (int k0 = 0; k0 < Kdim; k0 += 32) {
    v16h a = frag_a_f32(A, aStride, m0, k0, lane);
#pragma unroll
    for (int i = 0; i < 4; ++i) {
      int nt = wave + i * 4;
      if (nt < nTiles) {  // wave-uniform
        v16h b = frag_b_f16(W16, Kdim, nt * 16, k0, lane);
        WMMA_F16(acc[i], a, b);
      }
    }
  }

  const int rowAdd = (lane >= 16) ? 8 : 0;
  const int colL = lane & 15;
#pragma unroll
  for (int i = 0; i < 4; ++i) {
    int nt = wave + i * 4;
    if (nt < nTiles) {
      int n = nt * 16 + colL;
      if (n < N) {
        float bv = (bias != nullptr) ? bias[n] : 0.0f;
#pragma unroll
        for (int r = 0; r < 8; ++r) {
          float v = acc[i][r] + bv;
          if (act == 1) v = v > 0.f ? v : 0.f;
          Out[(long)(m0 + r + rowAdd) * oStride + n] = v;
        }
      }
    }
  }
}

// ---------------------------- weight pre-conversion -------------------------
__global__ void rnng_f32_to_f16(const float* __restrict__ in, _Float16* __restrict__ out, long n) {
  long i = (long)blockIdx.x * 256 + threadIdx.x;
  if (i < n) out[i] = (_Float16)in[i];
}
__global__ void rnng_f32_to_f16_padN(const float* __restrict__ in, _Float16* __restrict__ out,
                                     int N, int Kd, long n) {  // n = Npad*Kd
  long i = (long)blockIdx.x * 256 + threadIdx.x;
  if (i >= n) return;
  long row = i / Kd;
  out[i] = (row < N) ? (_Float16)in[i] : (_Float16)0.0f;
}

// ---------------------------- elementwise helpers ---------------------------
__global__ void rnng_zero(float* p, long n) {
  long i = (long)blockIdx.x * 256 + threadIdx.x;
  if (i < n) p[i] = 0.f;
}

__global__ void rnng_bcast_state(const float* __restrict__ src, float* __restrict__ dst,
                                 long rowStride, long n) {  // n = M*L*H
  long i = (long)blockIdx.x * 256 + threadIdx.x;
  if (i >= n) return;
  long row = i >> 9;
  int c = (int)(i & 511);
  dst[row * rowStride + c] = src[c];
}

__global__ void rnng_gather_wp(const int* __restrict__ words, const int* __restrict__ pos,
                               const float* __restrict__ wE, const float* __restrict__ pE,
                               float* __restrict__ X, long n) {  // n = B*TW*160
  long i = (long)blockIdx.x * 256 + threadIdx.x;
  if (i >= n) return;
  long row = i / 160;
  int c = (int)(i % 160);
  X[i] = (c < WDD) ? wE[(long)words[row] * WDD + c]
                   : pE[(long)pos[row] * PDD + (c - WDD)];
}

__global__ void rnng_gather_nt(const int* __restrict__ nt_ids, const float* __restrict__ ntE,
                               float* __restrict__ X, long n) {  // n = B*S*128
  long i = (long)blockIdx.x * 256 + threadIdx.x;
  if (i >= n) return;
  long row = i >> 7;
  int c = (int)(i & 127);
  X[i] = ntE[(long)nt_ids[row] * NTDD + c];
}

__global__ void rnng_gather_act(const int* __restrict__ nt_ids, const float* __restrict__ actE,
                                float* __restrict__ X, long n) {  // n = B*T*64
  long i = (long)blockIdx.x * 256 + threadIdx.x;
  if (i >= n) return;
  long row = i >> 6;
  int c = (int)(i & 63);
  int b = (int)(row / TT);
  int t = (int)(row % TT);
  int s = t / (KK + 2), p = t % (KK + 2);
  int idx = (p == 0) ? (nt_ids[b * SS + s] + 2) : ((p == KK + 1) ? 1 : 0);
  X[i] = actE[(long)idx * ADD_ + c];
}

__global__ void rnng_cat_fb(const float* __restrict__ fwdH, const float* __restrict__ bwdH,
                            float* __restrict__ cat, long n) {  // n = 2048*512
  long i = (long)blockIdx.x * 256 + threadIdx.x;
  if (i >= n) return;
  long row = i >> 9;
  int d = (int)(i & 511);
  cat[i] = (d < HH) ? fwdH[row * (LL * HH) + HH + d]
                    : bwdH[row * (LL * HH) + HH + (d - HH)];
}

__global__ void rnng_build_push(const float* __restrict__ nt_in, const float* __restrict__ xw,
                                const float* __restrict__ composed, float* __restrict__ push,
                                long n) {  // push: (T,B,D) row t*B+b
  long i = (long)blockIdx.x * 256 + threadIdx.x;
  if (i >= n) return;
  int d = (int)(i & 255);
  long rb = i >> 8;
  int b = (int)(rb % BB);
  int t = (int)(rb / BB);
  int s = t / (KK + 2), p = t % (KK + 2);
  float v;
  if (p == 0)       v = nt_in[((long)b * SS + s) * DD + d];
  else if (p <= KK) v = xw[((long)b * TWW + s * KK + (p - 1)) * DD + d];
  else              v = composed[((long)b * SS + s) * DD + d];
  push[i] = v;
}

__global__ void rnng_build_feats(const float* __restrict__ stack_tops,
                                 const float* __restrict__ o_buf,
                                 const float* __restrict__ hist_tops,
                                 float* __restrict__ feats, long n) {  // n = T*B*768
  long i = (long)blockIdx.x * 256 + threadIdx.x;
  if (i >= n) return;
  int d = (int)(i % 768);
  long row = i / 768;  // = t*B + b
  int b = (int)(row % BB);
  int t = (int)(row / BB);
  int s = t / (KK + 2), p = t % (KK + 2);
  int sh = (p >= 1) ? ((p - 1 < KK) ? (p - 1) : KK) : 0;
  int bufI = TWW - (s * KK + sh);
  float v;
  if (d < HH)          v = stack_tops[row * HH + d];
  else if (d < 2 * HH) v = o_buf[((long)bufI * BB + b) * HH + (d - HH)];
  else                 v = hist_tops[row * HH + (d - 2 * HH)];
  feats[i] = v;
}

__global__ void rnng_logsoftmax_t(const float* __restrict__ logits, float* __restrict__ out,
                                  long nrows) {  // rows = t*B+b ; out (B,T,A)
  long row = (long)blockIdx.x * 256 + threadIdx.x;
  if (row >= nrows) return;
  int b = (int)(row % BB);
  int t = (int)(row / BB);
  const float* L = logits + row * AAN;
  float mx = -3.0e38f;
  for (int a = 0; a < AAN; ++a) mx = L[a] > mx ? L[a] : mx;
  float sum = 0.f;
  for (int a = 0; a < AAN; ++a) sum += __expf(L[a] - mx);
  float ls = __logf(sum);
  float* o = out + ((long)b * TT + t) * AAN;
  for (int a = 0; a < AAN; ++a) o[a] = L[a] - mx - ls;
}

// ===========================================================================
extern "C" void kernel_launch(void* const* d_in, const int* in_sizes, int n_in,
                              void* d_out, int out_size, void* d_ws, size_t ws_size,
                              hipStream_t stream) {
  (void)in_sizes; (void)n_in; (void)out_size; (void)ws_size;
  // -------- inputs (setup_inputs order) --------
  const int*   words  = (const int*)d_in[0];
  const int*   pos    = (const int*)d_in[1];
  const int*   nt_ids = (const int*)d_in[2];
  const float* sW_ih  = (const float*)d_in[3];
  const float* sW_hh  = (const float*)d_in[4];
  const float* sb     = (const float*)d_in[5];
  const float* sh0    = (const float*)d_in[6];
  const float* sc0    = (const float*)d_in[7];
  const float* guards = (const float*)d_in[8];
  const float* cW_ih  = (const float*)d_in[9];
  const float* cW_hh  = (const float*)d_in[10];
  const float* cb     = (const float*)d_in[11];
  const float* w2l_W  = (const float*)d_in[12];  const float* w2l_b  = (const float*)d_in[13];
  const float* nt2l_W = (const float*)d_in[14];  const float* nt2l_b = (const float*)d_in[15];
  const float* a2l_W  = (const float*)d_in[16];  const float* a2l_b  = (const float*)d_in[17];
  const float* c2f_W  = (const float*)d_in[18];  const float* c2f_b  = (const float*)d_in[19];
  const float* l2s_W  = (const float*)d_in[20];  const float* l2s_b  = (const float*)d_in[21];
  const float* s2a_W  = (const float*)d_in[22];  const float* s2a_b  = (const float*)d_in[23];
  const float* word_E = (const float*)d_in[24];
  const float* pos_E  = (const float*)d_in[25];
  const float* nt_E   = (const float*)d_in[26];
  const float* act_E  = (const float*)d_in[27];
  float* out = (float*)d_out;

  // -------- workspace layout: f32 region then f16 region --------
  float* base = (float*)d_ws;
  size_t off = 0;
  auto alloc = [&](size_t n) { float* p = base + off; off += n; return p; };
  float* Xwp        = alloc((size_t)BB * TWW * 160);
  float* xw         = alloc((size_t)BB * TWW * DD);
  float* Xnt        = alloc((size_t)BB * SS * NTDD);
  float* nt_in      = alloc((size_t)BB * SS * DD);
  float* Xact       = alloc((size_t)BB * TT * ADD_);
  float* act_in     = alloc((size_t)BB * TT * DD);
  float* bufH       = alloc((size_t)BB * LL * HH);
  float* bufC       = alloc((size_t)BB * LL * HH);
  float* o_buf      = alloc((size_t)(TWW + 1) * BB * HH);
  float* histH      = alloc((size_t)BB * LL * HH);
  float* histC      = alloc((size_t)BB * LL * HH);
  float* hist_tops  = alloc((size_t)TT * BB * HH);
  float* fwdH       = alloc((size_t)BB * SS * LL * HH);
  float* fwdC       = alloc((size_t)BB * SS * LL * HH);
  float* bwdH       = alloc((size_t)BB * SS * LL * HH);
  float* bwdC       = alloc((size_t)BB * SS * LL * HH);
  float* catFB      = alloc((size_t)BB * SS * 2 * HH);
  float* composed   = alloc((size_t)BB * SS * DD);
  float* push_t     = alloc((size_t)TT * BB * DD);
  float* h_buf      = alloc((size_t)BB * DMAXX * LL * HH);
  float* c_buf      = alloc((size_t)BB * DMAXX * LL * HH);
  float* st0H       = alloc((size_t)BB * LL * HH);
  float* st0C       = alloc((size_t)BB * LL * HH);
  float* stack_tops = alloc((size_t)TT * BB * HH);
  float* feats      = alloc((size_t)TT * BB * 3 * HH);
  float* summary    = alloc((size_t)TT * BB * HH);
  float* logits     = alloc((size_t)TT * BB * AAN);

  _Float16* hb = (_Float16*)(base + off);
  size_t hoff = 0;
  auto halloc = [&](size_t n) { _Float16* p = hb + hoff; hoff += n; return p; };
  _Float16* sW_ih16  = halloc((size_t)3 * LL * GG * DD);
  _Float16* sW_hh16  = halloc((size_t)3 * LL * GG * HH);
  _Float16* cW_ih16  = halloc((size_t)2 * LL * GG * DD);
  _Float16* cW_hh16  = halloc((size_t)2 * LL * GG * HH);
  _Float16* w2l16    = halloc((size_t)DD * 160);
  _Float16* nt2l16   = halloc((size_t)DD * NTDD);
  _Float16* a2l16    = halloc((size_t)DD * ADD_);
  _Float16* c2f16    = halloc((size_t)DD * 2 * HH);
  _Float16* l2s16    = halloc((size_t)HH * 3 * HH);
  _Float16* s2a16    = halloc((size_t)AAPAD * HH);  // padded 80x256

  const long LH = (long)LL * HH;                   // 512
  const long stackStride = (long)DMAXX * LL * HH;  // 11776

  auto grid1 = [](long n) { return dim3((unsigned)((n + 255) / 256)); };
  auto cvt = [&](const float* src, _Float16* dst, long n) {
    rnng_f32_to_f16<<<grid1(n), 256, 0, stream>>>(src, dst, n);
  };

  auto lstm = [&](const float* x, long xs,
                  const _Float16* Wih, const _Float16* Whh, const float* bias,
                  const float* hIn, long his, const float* cIn, long cis,
                  float* hOut, long hos, float* cOut, long cos,
                  float* outLast, float* topOut, const float* topSrc, long ts, int M) {
    rnng_lstm_step<<<dim3(M / 16), dim3(128), 0, stream>>>(
        x, xs, Wih, Whh, bias, hIn, his, cIn, cis, hOut, hos, cOut, cos,
        outLast, topOut, topSrc, ts);
  };
  auto gemm = [&](const float* A, long as, int Kd, const _Float16* W16, int N, int Npad,
                  const float* bias, float* Out, long os, int act, int M) {
    rnng_gemm_bias_act<<<dim3(M / 16), dim3(128), 0, stream>>>(A, as, Kd, W16, N, Npad, bias, Out, os, act);
  };

  // -------- stack-pointer schedule (host replica of _schedule) --------
  int TOP_I[TT], READ_I[TT], WRITE_I[TT];
  {
    int ptr = 0, idx = 0;
    for (int s = 0; s < SS; ++s)
      for (int p = 0; p < KK + 2; ++p) {
        TOP_I[idx] = ptr;
        int pop = (p == KK + 1) ? (KK + 1) : 0;
        int r = ptr - pop;
        READ_I[idx] = r; WRITE_I[idx] = r + 1; ptr = r + 1;
        idx++;
      }
  }

  // -------- 0a. weight pre-conversion to f16 (static per launch) --------
  cvt(sW_ih, sW_ih16, (long)3 * LL * GG * DD);
  cvt(sW_hh, sW_hh16, (long)3 * LL * GG * HH);
  cvt(cW_ih, cW_ih16, (long)2 * LL * GG * DD);
  cvt(cW_hh, cW_hh16, (long)2 * LL * GG * HH);
  cvt(w2l_W, w2l16, (long)DD * 160);
  cvt(nt2l_W, nt2l16, (long)DD * NTDD);
  cvt(a2l_W, a2l16, (long)DD * ADD_);
  cvt(c2f_W, c2f16, (long)DD * 2 * HH);
  cvt(l2s_W, l2s16, (long)HH * 3 * HH);
  { long n = (long)AAPAD * HH;
    rnng_f32_to_f16_padN<<<grid1(n), 256, 0, stream>>>(s2a_W, s2a16, AAN, HH, n); }

  // -------- 0b. zero / broadcast initial states --------
  { long n = (long)BB * DMAXX * LL * HH;
    rnng_zero<<<grid1(n), 256, 0, stream>>>(h_buf, n);
    rnng_zero<<<grid1(n), 256, 0, stream>>>(c_buf, n); }
  { long n = (long)BB * SS * LL * HH;
    rnng_zero<<<grid1(n), 256, 0, stream>>>(fwdH, n);
    rnng_zero<<<grid1(n), 256, 0, stream>>>(fwdC, n);
    rnng_zero<<<grid1(n), 256, 0, stream>>>(bwdH, n);
    rnng_zero<<<grid1(n), 256, 0, stream>>>(bwdC, n); }
  { long n = (long)BB * LL * HH;
    rnng_bcast_state<<<grid1(n), 256, 0, stream>>>(sh0 + 1 * LH, bufH, LH, n);
    rnng_bcast_state<<<grid1(n), 256, 0, stream>>>(sc0 + 1 * LH, bufC, LH, n);
    rnng_bcast_state<<<grid1(n), 256, 0, stream>>>(sh0 + 2 * LH, histH, LH, n);
    rnng_bcast_state<<<grid1(n), 256, 0, stream>>>(sc0 + 2 * LH, histC, LH, n);
    rnng_bcast_state<<<grid1(n), 256, 0, stream>>>(sh0 + 0 * LH, st0H, LH, n);
    rnng_bcast_state<<<grid1(n), 256, 0, stream>>>(sc0 + 0 * LH, st0C, LH, n); }

  // -------- 1. input projections --------
  { long n = (long)BB * TWW * 160;
    rnng_gather_wp<<<grid1(n), 256, 0, stream>>>(words, pos, word_E, pos_E, Xwp, n); }
  gemm(Xwp, 160, 160, w2l16, DD, DD, w2l_b, xw, DD, 1, BB * TWW);
  { long n = (long)BB * SS * NTDD;
    rnng_gather_nt<<<grid1(n), 256, 0, stream>>>(nt_ids, nt_E, Xnt, n); }
  gemm(Xnt, NTDD, NTDD, nt2l16, DD, DD, nt2l_b, nt_in, DD, 1, BB * SS);
  { long n = (long)BB * TT * ADD_;
    rnng_gather_act<<<grid1(n), 256, 0, stream>>>(nt_ids, act_E, Xact, n); }
  gemm(Xact, ADD_, ADD_, a2l16, DD, DD, a2l_b, act_in, DD, 1, BB * TT);

  const long wStride = (long)LL * GG * DD;  // per-LSTM f16 weight block
  const long bStride = (long)LL * GG;

  // -------- 2. buffer LSTM (65 steps) --------
  for (int i = 0; i <= TWW; ++i) {
    const float* x = (i == 0) ? (guards + 1 * DD) : (xw + (long)(TWW - i) * DD);
    long xs = (i == 0) ? 0 : (long)TWW * DD;
    lstm(x, xs, sW_ih16 + 1 * wStride, sW_hh16 + 1 * wStride, sb + 1 * bStride,
         bufH, LH, bufC, LH, bufH, LH, bufC, LH,
         o_buf + (long)i * BB * HH, nullptr, nullptr, 0, BB);
  }

  // -------- 3. history LSTM (96 steps) --------
  for (int i = 0; i < TT; ++i) {
    const float* x = (i == 0) ? (guards + 2 * DD) : (act_in + (long)(i - 1) * DD);
    long xs = (i == 0) ? 0 : (long)TT * DD;
    lstm(x, xs, sW_ih16 + 2 * wStride, sW_hh16 + 2 * wStride, sb + 2 * bStride,
         histH, LH, histC, LH, histH, LH, histC, LH,
         hist_tops + (long)i * BB * HH, nullptr, nullptr, 0, BB);
  }

  // -------- 4. compose LSTMs (fwd/bwd, 5 steps each, M = B*S) --------
  for (int t = 0; t <= KK; ++t) {
    const float* xf = (t == 0) ? nt_in : (xw + (long)(t - 1) * DD);
    long xfs = (t == 0) ? (long)DD : (long)KK * DD;
    lstm(xf, xfs, cW_ih16 + 0 * wStride, cW_hh16 + 0 * wStride, cb + 0 * bStride,
         fwdH, LH, fwdC, LH, fwdH, LH, fwdC, LH, nullptr, nullptr, nullptr, 0, BB * SS);
    const float* xb = (t == 0) ? nt_in : (xw + (long)(KK - t) * DD);
    long xbs = (t == 0) ? (long)DD : (long)KK * DD;
    lstm(xb, xbs, cW_ih16 + 1 * wStride, cW_hh16 + 1 * wStride, cb + 1 * bStride,
         bwdH, LH, bwdC, LH, bwdH, LH, bwdC, LH, nullptr, nullptr, nullptr, 0, BB * SS);
  }
  { long n = (long)BB * SS * 2 * HH;
    rnng_cat_fb<<<grid1(n), 256, 0, stream>>>(fwdH, bwdH, catFB, n); }
  gemm(catFB, 2 * HH, 2 * HH, c2f16, DD, DD, c2f_b, composed, DD, 1, BB * SS);

  // -------- 5. push sequence + stack scan --------
  { long n = (long)TT * BB * DD;
    rnng_build_push<<<grid1(n), 256, 0, stream>>>(nt_in, xw, composed, push_t, n); }
  lstm(guards + 0 * DD, 0, sW_ih16, sW_hh16, sb,
       st0H, LH, st0C, LH,
       h_buf, stackStride, c_buf, stackStride,
       nullptr, nullptr, nullptr, 0, BB);
  for (int t = 0; t < TT; ++t) {
    int ri = READ_I[t], wi = WRITE_I[t], ti = TOP_I[t];
    lstm(push_t + (long)t * BB * DD, DD, sW_ih16, sW_hh16, sb,
         h_buf + (long)ri * LH, stackStride, c_buf + (long)ri * LH, stackStride,
         h_buf + (long)wi * LH, stackStride, c_buf + (long)wi * LH, stackStride,
         nullptr, stack_tops + (long)t * BB * HH,
         h_buf + (long)ti * LH + (LL - 1) * HH, stackStride, BB);
  }

  // -------- 6. head --------
  { long n = (long)TT * BB * 3 * HH;
    rnng_build_feats<<<grid1(n), 256, 0, stream>>>(stack_tops, o_buf, hist_tops, feats, n); }
  gemm(feats, 3 * HH, 3 * HH, l2s16, HH, HH, l2s_b, summary, HH, 1, TT * BB);
  gemm(summary, HH, HH, s2a16, AAN, AAPAD, s2a_b, logits, AAN, 0, TT * BB);
  { long n = (long)TT * BB;
    rnng_logsoftmax_t<<<grid1(n), 256, 0, stream>>>(logits, out, n); }
}
